// DistSelfAttention_87084756894192
// MI455X (gfx1250) — matrix operand order repaired and checked by reading, not compiled, not run on hardware
//
#include <hip/hip_runtime.h>
#include <hip/hip_bf16.h>
#include <math.h>

#define BB 2
#define SS 1024
#define DD 1024
#define HH 16
#define HD 64
#define EPSF 1e-6f

#define GM (BB * SS)   // 2048
#define GN DD          // 1024
#define GK DD          // 1024

typedef _Float16 v8h  __attribute__((ext_vector_type(8)));
typedef _Float16 v16h __attribute__((ext_vector_type(16)));
typedef float    v8f  __attribute__((ext_vector_type(8)));

// ---------------------------------------------------------------------------
// WMMA helpers (CDNA5 16x16x32 f16 -> f32)
// ---------------------------------------------------------------------------
__device__ __forceinline__ v8f wmma_f16(v16h a, v16h b, v8f c) {
  return __builtin_amdgcn_wmma_f32_16x16x32_f16(false, a, false, b, (short)0, c,
                                                false, false);
}

// Assemble A-frag from a hoisted per-lane pointer: elems[0..7] = p[0..7],
// elems[8..15] = p[8..15]  (pointer already includes row, k_base, hi*16)
__device__ __forceinline__ v16h a_frag_at(const _Float16* p) {
  v8h lo = *(const v8h*)p;
  v8h hh = *(const v8h*)(p + 8);
  v16h f;
#pragma unroll
  for (int i = 0; i < 8; ++i) { f[i] = lo[i]; f[i + 8] = hh[i]; }
  return f;
}

// A-frag (16x32, MxK): lane l -> row r=l&15; elems[0..15]=A[r][k+hi*16..+15]
//   (hi = l>>4)
__device__ __forceinline__ v16h load_a_frag(const _Float16* base, int ld,
                                            int row_base, int k_base, int lane) {
  int r = lane & 15, hi = (lane >> 4) & 1;
  return a_frag_at(base + (size_t)(row_base + r) * ld + k_base + hi * 16);
}

// B-frag for NT gemm: B[k][n] = W[n][k]; lane l -> col n=l&15;
// elems[0..15] = W[n][k+hi*16 .. +15]  (contiguous row chunk)
__device__ __forceinline__ v16h load_b_frag_nt(const _Float16* base, int ld,
                                               int n_base, int k_base, int lane) {
  int n = lane & 15, hi = (lane >> 4) & 1;
  const _Float16* p = base + (size_t)(n_base + n) * ld + k_base + hi * 16;
  return *(const v16h*)p;
}

// ---------------------------------------------------------------------------
// elementwise convert kernels
// ---------------------------------------------------------------------------
__global__ void cvt_f16_kernel(const float* __restrict__ in,
                               _Float16* __restrict__ out, int n, int square) {
  int i = blockIdx.x * blockDim.x + threadIdx.x;
  if (i >= n) return;
  float v = in[i];
  if (square) v *= v;
  out[i] = (_Float16)v;
}

// ---------------------------------------------------------------------------
// NT GEMM: C[GM,GN] = A[GM,GK] * W[GN,GK]^T  (+epilogue). wave tile 32M x 64N
// ---------------------------------------------------------------------------
#define EP_BIAS_F16 0
#define EP_VVAR_F16 1
#define EP_BIAS_F32 2
#define EP_SQRT_F32 3

__global__ void
__launch_bounds__(256, 1)
gemm_nt_kernel(const _Float16* __restrict__ A, const _Float16* __restrict__ W,
               const float* __restrict__ bias, void* out, int ep) {
  const int lane = threadIdx.x & 31;
  const int wave = (blockIdx.x * blockDim.x + threadIdx.x) >> 5;
  const int tiles_n = GN >> 6;
  const int tm = wave / tiles_n;
  const int tn = wave - tm * tiles_n;
  const int m0 = tm * 32, n0 = tn * 64;
  if (m0 >= GM) return;

  v8f acc[2][4];
#pragma unroll
  for (int rb = 0; rb < 2; ++rb)
#pragma unroll
    for (int i = 0; i < 4; ++i) acc[rb][i] = v8f{};

  // hoisted per-lane fragment base pointers (k advances via index only)
  const int r = lane & 15;
  const int hi = (lane >> 4) & 1;
  const _Float16* pa0 = A + (size_t)(m0 + r) * GK + hi * 16;
  const _Float16* pa1 = A + (size_t)(m0 + 16 + r) * GK + hi * 16;
  const _Float16* pb0 = W + (size_t)(n0 + r) * GK + hi * 16;
  const _Float16* pb1 = W + (size_t)(n0 + 16 + r) * GK + hi * 16;
  const _Float16* pb2 = W + (size_t)(n0 + 32 + r) * GK + hi * 16;
  const _Float16* pb3 = W + (size_t)(n0 + 48 + r) * GK + hi * 16;

  __builtin_prefetch((const void*)pa0, 0, 1);
  __builtin_prefetch((const void*)pa1, 0, 1);

#pragma unroll 4
  for (int k = 0; k < GK; k += 32) {
    v16h a0 = a_frag_at(pa0 + k);
    v16h a1 = a_frag_at(pa1 + k);
    v16h b0 = *(const v16h*)(pb0 + k);
    v16h b1 = *(const v16h*)(pb1 + k);
    v16h b2 = *(const v16h*)(pb2 + k);
    v16h b3 = *(const v16h*)(pb3 + k);

    acc[0][0] = wmma_f16(a0, b0, acc[0][0]);
    acc[1][0] = wmma_f16(a1, b0, acc[1][0]);
    acc[0][1] = wmma_f16(a0, b1, acc[0][1]);
    acc[1][1] = wmma_f16(a1, b1, acc[1][1]);
    acc[0][2] = wmma_f16(a0, b2, acc[0][2]);
    acc[1][2] = wmma_f16(a1, b2, acc[1][2]);
    acc[0][3] = wmma_f16(a0, b3, acc[0][3]);
    acc[1][3] = wmma_f16(a1, b3, acc[1][3]);
  }

  const int nn = lane & 15;
#pragma unroll
  for (int rb = 0; rb < 2; ++rb) {
#pragma unroll
    for (int i = 0; i < 4; ++i) {
      int ncol = n0 + i * 16 + nn;
      float bv = (bias != nullptr) ? bias[ncol] : 0.0f;
#pragma unroll
      for (int j = 0; j < 8; ++j) {
        int row = m0 + rb * 16 + j + 8 * hi;
        float v = acc[rb][i][j];
        size_t idx = (size_t)row * GN + ncol;
        if (ep == EP_BIAS_F16) {
          ((_Float16*)out)[idx] = (_Float16)(v + bv);
        } else if (ep == EP_VVAR_F16) {
          float s = sqrtf(fmaxf(v, 0.0f)) + EPSF;   // vvar = (sqrt(t)+eps)^2
          ((_Float16*)out)[idx] = (_Float16)(s * s);
        } else if (ep == EP_BIAS_F32) {
          ((float*)out)[idx] = v + bv;
        } else {
          ((float*)out)[idx] = sqrtf(fmaxf(v, 0.0f));
        }
      }
    }
  }
}

// ---------------------------------------------------------------------------
// transpose [B,S,D] f16 -> [B,H,HD,S] f16
// ---------------------------------------------------------------------------
__global__ void transpose_kernel(const _Float16* __restrict__ in,
                                 _Float16* __restrict__ out) {
  int i = blockIdx.x * blockDim.x + threadIdx.x;
  if (i >= BB * SS * DD) return;
  int dfull = i % DD;
  int t = (i / DD) % SS;
  int b = i / (DD * SS);
  int h = dfull / HD, d = dfull % HD;
  out[(((size_t)b * HH + h) * HD + d) * SS + t] = in[i];
}

// colsum over t of vvar per (b,h,d): [B,H,HD] f32
__global__ void colsum_kernel(const _Float16* __restrict__ vvar,
                              float* __restrict__ out) {
  int i = blockIdx.x * blockDim.x + threadIdx.x;
  if (i >= BB * HH * HD) return;
  int d = i % HD;
  int h = (i / HD) % HH;
  int b = i / (HD * HH);
  float s = 0.0f;
  for (int t = 0; t < SS; ++t)
    s += (float)vvar[((size_t)b * SS + t) * DD + h * HD + d];
  out[i] = s;
}

// ---------------------------------------------------------------------------
// flash attention (dual-softmax distributional). 1 wave / block, 16 q-rows.
// ---------------------------------------------------------------------------
__global__ void
__launch_bounds__(32)
attn_kernel(const _Float16* __restrict__ qm, const _Float16* __restrict__ km,
            const _Float16* __restrict__ vmT, const _Float16* __restrict__ vvT,
            const float* __restrict__ colsum, const float* __restrict__ tau_p,
            _Float16* __restrict__ ymu, _Float16* __restrict__ x2) {
  __shared__ float sc[16][32];
  __shared__ float st_m[16], st_z1[16], st_z2[16];
  __shared__ float st_r1[16], st_r2[16], st_r3[16];

  const int lane = threadIdx.x;
  const int blk = blockIdx.x;
  const int stile = blk & (SS / 16 - 1);
  const int h = (blk / (SS / 16)) % HH;
  const int b = blk / ((SS / 16) * HH);
  const int s0 = stile * 16;

  const float tau = tau_p[0];
  const float scl1 = 1.0f / (8.0f * tau);                 // l_mu = raw * scl1
  float l_var = (0.1f + EPSF) / (float)HD + EPSF;
  l_var = l_var / (tau * tau) + EPSF;
  const float cadj = rsqrtf(1.0f + (float)(M_PI / 8.0) * l_var);
  const float scl2 = cadj * scl1;

  const _Float16* qb = qm + (size_t)b * SS * DD + h * HD;
  const _Float16* kb = km + (size_t)b * SS * DD + h * HD;
  const _Float16* vmb = vmT + ((size_t)b * HH + h) * HD * SS;
  const _Float16* vvb = vvT + ((size_t)b * HH + h) * HD * SS;

  v8f a1[4] = {v8f{}, v8f{}, v8f{}, v8f{}};
  v8f a2[4] = {v8f{}, v8f{}, v8f{}, v8f{}};
  v8f aB[4] = {v8f{}, v8f{}, v8f{}, v8f{}};
  v8f aC[4] = {v8f{}, v8f{}, v8f{}, v8f{}};

  if (lane < 16) { st_m[lane] = -1e30f; st_z1[lane] = 0.f; st_z2[lane] = 0.f; }
  __syncthreads();

  const int hi = (lane >> 4) & 1;
  const int nl = lane & 15;

  for (int t0 = 0; t0 < SS; t0 += 32) {
    // raw score tiles: [16 q-rows] x [32 keys], K-dim = HD = 64
    v8f sf0 = v8f{}, sf1 = v8f{};
#pragma unroll
    for (int kbase = 0; kbase < HD; kbase += 32) {
      v16h aq = load_a_frag(qb, DD, s0, kbase, lane);
      v16h b0 = load_b_frag_nt(kb, DD, t0, kbase, lane);
      v16h b1 = load_b_frag_nt(kb, DD, t0 + 16, kbase, lane);
      sf0 = wmma_f16(aq, b0, sf0);
      sf1 = wmma_f16(aq, b1, sf1);
    }

    __syncthreads();  // previous tile's sc reads are done
#pragma unroll
    for (int j = 0; j < 8; ++j) {
      sc[j + 8 * hi][nl] = sf0[j];
      sc[j + 8 * hi][nl + 16] = sf1[j];
    }
    __syncthreads();

    if (lane < 16) {
      float mold = st_m[lane];
      float mnew = mold;
      for (int t = 0; t < 32; ++t) mnew = fmaxf(mnew, sc[lane][t]);
      float r1 = __expf((mold - mnew) * scl1);
      float r2 = __expf((mold - mnew) * scl2);
      float z1 = st_z1[lane] * r1, z2 = st_z2[lane] * r2;
      for (int t = 0; t < 32; ++t) {
        float d = sc[lane][t] - mnew;
        z1 += __expf(d * scl1);
        z2 += __expf(d * scl2);
      }
      st_m[lane] = mnew; st_z1[lane] = z1; st_z2[lane] = z2;
      st_r1[lane] = r1; st_r2[lane] = r2; st_r3[lane] = r2 * r2;
    }
    __syncthreads();

    // rescale running accumulators (per output row = j + 8*hi)
#pragma unroll
    for (int j = 0; j < 8; ++j) {
      int row = j + 8 * hi;
      float f1 = st_r1[row], f2 = st_r2[row], f3 = st_r3[row];
#pragma unroll
      for (int i = 0; i < 4; ++i) {
        a1[i][j] *= f1; a2[i][j] *= f2; aB[i][j] *= f2; aC[i][j] *= f3;
      }
    }

    // build probability A-frags for this 16x32 tile
    const int rr = lane & 15;
    const float mrow = st_m[rr];
    v16h p1, p2, p3;
#pragma unroll
    for (int e = 0; e < 16; ++e) {
      int col = hi * 16 + e;
      float d = sc[rr][col] - mrow;
      float e1 = __expf(d * scl1);
      float e2 = __expf(d * scl2);
      p1[e] = (_Float16)e1;
      p2[e] = (_Float16)e2;
      p3[e] = (_Float16)(e2 * e2);
    }

    // PV accumulation: B operands from transposed vm / vvar ([d][t] rows)
#pragma unroll
    for (int i = 0; i < 4; ++i) {
      v16h bvm = load_b_frag_nt(vmb, SS, i * 16, t0, lane);
      v16h bvv = load_b_frag_nt(vvb, SS, i * 16, t0, lane);
      a1[i] = wmma_f16(p1, bvm, a1[i]);
      a2[i] = wmma_f16(p2, bvm, a2[i]);
      aB[i] = wmma_f16(p2, bvv, aB[i]);
      aC[i] = wmma_f16(p3, bvv, aC[i]);
    }
  }

  const float base_c = 1e-4f + EPSF;  // top_var + eps (constant part of attn_var)
#pragma unroll
  for (int i = 0; i < 4; ++i) {
    int d = i * 16 + nl;
    float cs = colsum[((size_t)b * HH + h) * HD + d];
#pragma unroll
    for (int j = 0; j < 8; ++j) {
      int row = j + 8 * hi;
      float Z1 = st_z1[row], Z2 = st_z2[row];
      float mu = a1[i][j] / Z1 + a2[i][j] / Z2;
      float var = base_c * cs + l_var * (aB[i][j] / Z2 - aC[i][j] / (Z2 * Z2));
      float ysc = sqrtf(fmaxf(var, 0.0f) + EPSF);
      float x2v = ysc + EPSF;
      x2v *= x2v;
      size_t idx = ((size_t)b * SS + (s0 + row)) * DD + h * HD + d;
      ymu[idx] = (_Float16)mu;
      x2[idx] = (_Float16)x2v;
    }
  }
}

// ---------------------------------------------------------------------------
// launch
// ---------------------------------------------------------------------------
extern "C" void kernel_launch(void* const* d_in, const int* in_sizes, int n_in,
                              void* d_out, int out_size, void* d_ws,
                              size_t ws_size, hipStream_t stream) {
  (void)in_sizes; (void)n_in; (void)out_size; (void)ws_size;

  const float* q_loc = (const float*)d_in[0];
  const float* k_loc = (const float*)d_in[2];
  const float* v_loc = (const float*)d_in[4];
  const float* v_scale = (const float*)d_in[5];
  const float* Wq = (const float*)d_in[6];
  const float* bq = (const float*)d_in[7];
  const float* Wk = (const float*)d_in[8];
  const float* bk = (const float*)d_in[9];
  const float* Wv = (const float*)d_in[10];
  const float* bv = (const float*)d_in[11];
  const float* Wo = (const float*)d_in[12];
  const float* bo = (const float*)d_in[13];
  const float* tau = (const float*)d_in[14];
  float* out = (float*)d_out;

  const size_t BSD = (size_t)BB * SS * DD;      // 2M elems
  const size_t WSZ = (size_t)DD * DD;           // 1M elems
  char* ws = (char*)d_ws;
  const size_t MiB = 1u << 20;

  // region A (reused): 4 x 4MiB
  _Float16* hq   = (_Float16*)(ws + 0 * MiB);
  _Float16* hk   = (_Float16*)(ws + 4 * MiB);
  _Float16* hv   = (_Float16*)(ws + 8 * MiB);
  _Float16* hvs2 = (_Float16*)(ws + 12 * MiB);
  // reuse after projections complete:
  _Float16* vmT  = hq;
  _Float16* vvT  = hk;
  _Float16* ymu  = hv;
  _Float16* x2b  = hvs2;
  // region W: 6 x 2MiB
  _Float16* hWq  = (_Float16*)(ws + 16 * MiB);
  _Float16* hWk  = (_Float16*)(ws + 18 * MiB);
  _Float16* hWv  = (_Float16*)(ws + 20 * MiB);
  _Float16* hWv2 = (_Float16*)(ws + 22 * MiB);
  _Float16* hWo  = (_Float16*)(ws + 24 * MiB);
  _Float16* hWo2 = (_Float16*)(ws + 26 * MiB);
  // region P (persists through attention): 4 x 4MiB
  _Float16* qm   = (_Float16*)(ws + 28 * MiB);
  _Float16* km   = (_Float16*)(ws + 32 * MiB);
  _Float16* vm   = (_Float16*)(ws + 36 * MiB);
  _Float16* vvar = (_Float16*)(ws + 40 * MiB);
  float* csum    = (float*)(ws + 44 * MiB);     // B*H*HD f32

  // 1) convert to f16 (activations + squared scale, weights + squared weights)
  {
    int na = (int)BSD, nw = (int)WSZ, blk = 256;
    cvt_f16_kernel<<<na / blk, blk, 0, stream>>>(q_loc, hq, na, 0);
    cvt_f16_kernel<<<na / blk, blk, 0, stream>>>(k_loc, hk, na, 0);
    cvt_f16_kernel<<<na / blk, blk, 0, stream>>>(v_loc, hv, na, 0);
    cvt_f16_kernel<<<na / blk, blk, 0, stream>>>(v_scale, hvs2, na, 1);
    cvt_f16_kernel<<<nw / blk, blk, 0, stream>>>(Wq, hWq, nw, 0);
    cvt_f16_kernel<<<nw / blk, blk, 0, stream>>>(Wk, hWk, nw, 0);
    cvt_f16_kernel<<<nw / blk, blk, 0, stream>>>(Wv, hWv, nw, 0);
    cvt_f16_kernel<<<nw / blk, blk, 0, stream>>>(Wv, hWv2, nw, 1);
    cvt_f16_kernel<<<nw / blk, blk, 0, stream>>>(Wo, hWo, nw, 0);
    cvt_f16_kernel<<<nw / blk, blk, 0, stream>>>(Wo, hWo2, nw, 1);
  }

  // 2) projection GEMMs: M=2048, N=K=1024, wave tile 32x64
  const int gblk = 256;
  const int gwaves = (GM / 32) * (GN / 64);
  const int ggrid = (gwaves * 32 + gblk - 1) / gblk;
  gemm_nt_kernel<<<ggrid, gblk, 0, stream>>>(hq, hWq, bq, qm, EP_BIAS_F16);
  gemm_nt_kernel<<<ggrid, gblk, 0, stream>>>(hk, hWk, bk, km, EP_BIAS_F16);
  gemm_nt_kernel<<<ggrid, gblk, 0, stream>>>(hv, hWv, bv, vm, EP_BIAS_F16);
  gemm_nt_kernel<<<ggrid, gblk, 0, stream>>>(hvs2, hWv2, nullptr, vvar, EP_VVAR_F16);

  // 3) transposes for PV B-operands + vvar column sums
  transpose_kernel<<<(int)(BSD / 256), 256, 0, stream>>>(vm, vmT);
  transpose_kernel<<<(int)(BSD / 256), 256, 0, stream>>>(vvar, vvT);
  colsum_kernel<<<(BB * HH * HD) / 256, 256, 0, stream>>>(vvar, csum);

  // 4) attention: one wave per 16 query rows per head
  attn_kernel<<<BB * HH * (SS / 16), 32, 0, stream>>>(qm, km, vmT, vvT, csum,
                                                      tau, ymu, x2b);

  // 5) output projections straight into d_out (loc then scale)
  gemm_nt_kernel<<<ggrid, gblk, 0, stream>>>(ymu, hWo, bo, out, EP_BIAS_F32);
  gemm_nt_kernel<<<ggrid, gblk, 0, stream>>>(x2b, hWo2, nullptr, out + BSD, EP_SQRT_F32);
}
